// PatchTokenizer3D_4080218931654
// MI455X (gfx1250) — compile-verified, matrix operand order checked by reading, not run on hardware
//
#include <hip/hip_runtime.h>
#include <hip/hip_bf16.h>

// ---------------------------------------------------------------------------
// CDNA5 / gfx1250 implementation of PatchTokenizer3D
//   images: (8,1,160,160,160) fp32
//   3 scales: 160 (patch grid 20^3), 80 (10^3), 40 (5^3), patch = 8^3
//   Per-patch std (ddof=1) via V_WMMA_F32_16X16X32_F16 row-sum reduction,
//   staged through LDS with GLOBAL_LOAD_ASYNC_TO_LDS_B128 (ASYNCcnt path).
//   Threshold test done on unnormalized SS: std>thr <=> SS > thr^2*(n-1).
// ---------------------------------------------------------------------------

typedef __attribute__((ext_vector_type(16))) _Float16 v16h;
typedef __attribute__((ext_vector_type(8)))  float    v8f;
typedef __attribute__((ext_vector_type(4)))  int      v4i;

#define IMG0 160
#define NP0  64000   // 8 * 20^3
#define NP1  8000    // 8 * 10^3
#define NP2  1000    // 8 * 5^3

// ---- CDNA5 async global->LDS copy (ASYNCcnt), with safe fallback ----------
// Builtin signature (from hipcc diagnostic): pointee type is int __vector(4).
#if defined(__HIP_DEVICE_COMPILE__) && __has_builtin(__builtin_amdgcn_global_load_async_to_lds_b128)
#define HAVE_ASYNC_LDS 1
#define ASYNC_COPY16(gsrc, ldst)                                              \
  __builtin_amdgcn_global_load_async_to_lds_b128(                             \
      (__attribute__((address_space(1))) v4i*)(gsrc),                         \
      (__attribute__((address_space(3))) v4i*)(ldst), 0, 0)
#if __has_builtin(__builtin_amdgcn_s_wait_asynccnt)
#define ASYNC_WAIT() __builtin_amdgcn_s_wait_asynccnt(0)
#else
#define ASYNC_WAIT() asm volatile("s_wait_asynccnt 0" ::: "memory")
#endif
#else
#define HAVE_ASYNC_LDS 0
#define ASYNC_COPY16(gsrc, ldst) (*(float4*)(ldst) = *(const float4*)(gsrc))
#define ASYNC_WAIT() ((void)0)
#endif

// -------------------------- downsample x2 (trilinear, align_corners=False =>
// exact 2x2x2 average) ------------------------------------------------------
__global__ __launch_bounds__(256)
void down2_kernel(const float* __restrict__ in, float* __restrict__ out) {
  int i = blockIdx.x * 256 + threadIdx.x;
  const int N = 8 * 80 * 80 * 80;
  if (i >= N) return;
  int x = i % 80; int t = i / 80;
  int y = t % 80; t /= 80;
  int z = t % 80; int b = t / 80;
  const float* p = in + (((size_t)b * 160 + 2 * z) * 160 + 2 * y) * 160 + 2 * x;
  __builtin_prefetch(p + 2 * 160 * 160, 0, 1);   // global_prefetch_b8
  float2 a0 = *(const float2*)(p);
  float2 a1 = *(const float2*)(p + 160);
  float2 a2 = *(const float2*)(p + 160 * 160);
  float2 a3 = *(const float2*)(p + 160 * 160 + 160);
  out[i] = (a0.x + a0.y + a1.x + a1.y + a2.x + a2.y + a3.x + a3.y) * 0.125f;
}

// -------------------------- downsample x4 (trilinear direct 160->40: sample
// at 4i+1.5 => average of taps {4i+1, 4i+2} per axis) ------------------------
__global__ __launch_bounds__(256)
void down4_kernel(const float* __restrict__ in, float* __restrict__ out) {
  int i = blockIdx.x * 256 + threadIdx.x;
  const int N = 8 * 40 * 40 * 40;
  if (i >= N) return;
  int x = i % 40; int t = i / 40;
  int y = t % 40; t /= 40;
  int z = t % 40; int b = t / 40;
  const float* p = in + (((size_t)b * 160 + 4 * z + 1) * 160 + 4 * y + 1) * 160 + 4 * x + 1;
  float s = 0.f;
#pragma unroll
  for (int dz = 0; dz < 2; ++dz)
#pragma unroll
    for (int dy = 0; dy < 2; ++dy) {
      const float* q = p + dz * 160 * 160 + dy * 160;
      s += q[0] + q[1];
    }
  out[i] = s * 0.125f;
}

// -------------------------- per-patch std via WMMA -------------------------
// One wave (32 lanes) handles 16 patches. Patch data (512 f32) staged in LDS
// via async global->LDS copies; f32->f16 conversion happens at fragment-build
// time (co-executes with 16-bit XDL WMMA, which is tracked as a TRANS op).
// Row sums: D = A x Ones + C with V_WMMA_F32_16X16X32_F16 (fp32 accumulate).
// thrSS = thr^2 * (n-1): mask = (sumsq - sum^2/n) > thrSS  (no div, no sqrt).
template <int S>
__global__ __launch_bounds__(32)
void patch_std_kernel(const float* __restrict__ src, int* __restrict__ mask,
                      float thrSS, int totalPatches) {
  const int G = S / 8;                 // patch grid per dim
  __shared__ float lds[16][516];       // +4 f32 row pad: row base rotates 4 banks
  __shared__ float s_sum[16];
  __shared__ float s_sq[16];

  const int lane  = threadIdx.x;
  const int pbase = blockIdx.x * 16;

  // cooperative staging: 16 patches * 128 float4 -> LDS (async path)
  for (int i = lane; i < 16 * 128; i += 32) {
    int p   = i >> 7;        // patch 0..15
    int r   = i & 127;       // float4 index within patch
    int row = r >> 1;        // (dz,hy) row 0..63
    int h   = r & 1;         // which half of the 8-float row
    int gp  = pbase + p;
    int e   = row * 8 + h * 4;
    if (gp < totalPatches) {
      int gx = gp % G; int t = gp / G;
      int gy = t % G;  t /= G;
      int gz = t % G;  int b = t / G;
      int dz = row >> 3, hy = row & 7;
      const float* q = src + (((size_t)b * S + (gz * 8 + dz)) * S + (gy * 8 + hy)) * S
                       + gx * 8 + h * 4;
      ASYNC_COPY16(q, &lds[p][e]);
    } else {
      *(float4*)&lds[p][e] = make_float4(0.f, 0.f, 0.f, 0.f);
    }
  }
  ASYNC_WAIT();
  __syncthreads();

  // B = all ones (column sums == full K-row sums regardless of layout)
  v16h ones;
#pragma unroll
  for (int j = 0; j < 16; ++j) ones[j] = (_Float16)1.0f;

  v8f csum = {};
  v8f csq  = {};
  const int M  = lane & 15;   // patch row of A
  const int hi = lane >> 4;   // half-K selector per ISA 16-bit A layout

  for (int k0 = 0; k0 < 512; k0 += 32) {
    // lanes 0-15 : K = k0..k0+7    and k0+16..k0+23
    // lanes 16-31: K = k0+8..k0+15 and k0+24..k0+31
    const float* pA = &lds[M][k0 + hi * 8];
    v16h a;
#pragma unroll
    for (int j = 0; j < 8; ++j) a[j] = (_Float16)pA[j];
#pragma unroll
    for (int j = 0; j < 8; ++j) a[8 + j] = (_Float16)pA[16 + j];
    v16h asq = a * a;   // packed f16 squares
    csum = __builtin_amdgcn_wmma_f32_16x16x32_f16(false, a,   false, ones,
                                                  (short)0, csum, false, false);
    csq  = __builtin_amdgcn_wmma_f32_16x16x32_f16(false, asq, false, ones,
                                                  (short)0, csq,  false, false);
  }

  // D layout: VGPR r holds M=r (lanes 0-15) / M=r+8 (lanes 16-31); every
  // column n is identical, so lane M==0 of each half publishes its rows.
  if (M == 0) {
#pragma unroll
    for (int r = 0; r < 8; ++r) {
      s_sum[r + hi * 8] = csum[r];
      s_sq [r + hi * 8] = csq[r];
    }
  }
  __syncthreads();

  if (lane < 16) {
    int gp = pbase + lane;
    if (gp < totalPatches) {
      float sum = s_sum[lane];
      float sq  = s_sq[lane];
      float ss  = fmaf(sum * sum, -(1.0f / 512.0f), sq);  // SS = sq - sum^2/n
      mask[gp] = (ss > thrSS) ? 1 : 0;
    }
  }
}

// -------------------------- exclusive scan (single workgroup) --------------
__global__ __launch_bounds__(1024)
void scan_kernel(const int* __restrict__ mask, int* __restrict__ offs, int n) {
  __shared__ int partial[1024];
  int t = threadIdx.x;
  int chunk = (n + 1023) / 1024;
  int beg = t * chunk;
  int end = beg + chunk; if (end > n) end = n;
  int s = 0;
  for (int i = beg; i < end && i < n; ++i) s += mask[i];
  partial[t] = s;
  __syncthreads();
  if (t == 0) {
    int acc = 0;
    for (int i = 0; i < 1024; ++i) { int v = partial[i]; partial[i] = acc; acc += v; }
  }
  __syncthreads();
  int acc = partial[t];
  for (int i = beg; i < end && i < n; ++i) { int m = mask[i]; offs[i] = acc; acc += m; }
}

// -------------------------- meta: counts, seqlens, section bases -----------
// meta[0..10] = float offsets of: r0,r1,r2,full1,full2,pm0,pm1,pm2,padded,
//               output_mask,cu_seqlens.  meta[11] = maxlen.
__global__ __launch_bounds__(256)
void meta_kernel(const int* __restrict__ m0, const int* __restrict__ o0,
                 const int* __restrict__ m1, const int* __restrict__ o1,
                 const int* __restrict__ m2, const int* __restrict__ o2,
                 int* __restrict__ meta, float* __restrict__ out) {
  __shared__ int cnt[8][3];
  __shared__ int seql[8];
  __shared__ int bases[12];
  __shared__ int s_ml;
  int t = threadIdx.x;
  const int PB[3] = {8000, 1000, 125};
  if (t < 24) {
    int b = t / 3, s = t % 3;
    const int* mm = (s == 0) ? m0 : (s == 1) ? m1 : m2;
    const int* oo = (s == 0) ? o0 : (s == 1) ? o1 : o2;
    int last = (b + 1) * PB[s] - 1;
    cnt[b][s] = (oo[last] + mm[last]) - oo[b * PB[s]];
  }
  __syncthreads();
  if (t == 0) {
    int k0 = 0, k1 = 0, k2 = 0, ml = 0;
    for (int b = 0; b < 8; ++b) {
      seql[b] = cnt[b][0] + cnt[b][1] + cnt[b][2];
      if (seql[b] > ml) ml = seql[b];
      k0 += cnt[b][0]; k1 += cnt[b][1]; k2 += cnt[b][2];
    }
    s_ml = ml;
    bases[0]  = 0;                          // resized0
    bases[1]  = k0 * 512;                   // resized1
    bases[2]  = (k0 + k1) * 512;            // resized2
    bases[3]  = (k0 + k1 + k2) * 512;       // full1
    bases[4]  = bases[3] + k1 * 4096;       // full2
    bases[5]  = bases[4] + k2 * 32768;      // posmask0
    bases[6]  = bases[5] + NP0;             // posmask1
    bases[7]  = bases[6] + NP1;             // posmask2
    bases[8]  = bases[7] + NP2;             // padded
    bases[9]  = bases[8] + 8 * ml;          // output_mask
    bases[10] = bases[9] + 8 * (ml + 1);    // cu_seqlens
    bases[11] = ml;
    for (int i = 0; i < 12; ++i) meta[i] = bases[i];
    // cu_seqlens (9 values, +1 CLS per sequence)
    float* cu = out + bases[10];
    int acc = 0; cu[0] = 0.f;
    for (int b = 0; b < 8; ++b) { acc += seql[b] + 1; cu[b + 1] = (float)acc; }
  }
  __syncthreads();
  int ml = s_ml;
  float* pad = out + bases[8];
  float* om  = out + bases[9];
  for (int b = 0; b < 8; ++b) {
    int c0 = cnt[b][0], c01 = c0 + cnt[b][1], c012 = c01 + cnt[b][2];
    for (int j = t; j < ml; j += blockDim.x) {
      float sval = (j < c0) ? 0.f : (j < c01) ? 1.f : (j < c012) ? 2.f : -1.f;
      pad[(size_t)b * ml + j] = sval;
      om[(size_t)b * (ml + 1) + 1 + j] = sval + 1.f;
    }
    if (t == 0) om[(size_t)b * (ml + 1)] = -1.f;
  }
}

// -------------------------- gather selected 8^3 patches --------------------
// Direct global->global float4 copy: both endpoints are HBM; staging through
// LDS would only add latency/LDS bandwidth. Roofline: whole pipeline ~280MB.
template <int S>
__global__ __launch_bounds__(64)
void gather_resized(const float* __restrict__ src, const int* __restrict__ mask,
                    const int* __restrict__ off, const int* __restrict__ meta,
                    int metaIdx, float* __restrict__ out) {
  const int G = S / 8;
  int p = blockIdx.x;
  if (!mask[p]) return;
  float* dst = out + meta[metaIdx] + (size_t)off[p] * 512;
  int gx = p % G; int t = p / G;
  int gy = t % G; t /= G;
  int gz = t % G; int b = t / G;
  for (int i = threadIdx.x; i < 128; i += 64) {
    int row = i >> 1, h = i & 1;
    int dz = row >> 3, hy = row & 7;
    float4 v = *(const float4*)(src + (((size_t)b * S + (gz * 8 + dz)) * S
                                       + (gy * 8 + hy)) * S + gx * 8 + h * 4);
    *(float4*)(dst + row * 8 + h * 4) = v;
  }
}

// -------------------------- gather full-res PS^3 patches -------------------
template <int PS>
__global__ __launch_bounds__(256)
void gather_full(const float* __restrict__ src, const int* __restrict__ mask,
                 const int* __restrict__ off, const int* __restrict__ meta,
                 int metaIdx, float* __restrict__ out) {
  const int G   = 160 / PS;
  const int NF4 = PS * PS * PS / 4;
  int p = blockIdx.x;
  if (!mask[p]) return;
  float* dst = out + meta[metaIdx] + (size_t)off[p] * (PS * PS * PS);
  int gx = p % G; int t = p / G;
  int gy = t % G; t /= G;
  int gz = t % G; int b = t / G;
  for (int i = threadIdx.x; i < NF4; i += 256) {
    int e  = i * 4;
    int wx = e % PS; int r = e / PS;
    int hy = r % PS; int dz = r / PS;
    float4 v = *(const float4*)(src + (((size_t)b * 160 + (gz * PS + dz)) * 160
                                       + (gy * PS + hy)) * 160 + gx * PS + wx);
    *(float4*)(dst + e) = v;
  }
}

// -------------------------- posmask (bool -> float) ------------------------
__global__ __launch_bounds__(256)
void posmask_kernel(const int* __restrict__ mask, int n,
                    const int* __restrict__ meta, int metaIdx,
                    float* __restrict__ out) {
  int i = blockIdx.x * 256 + threadIdx.x;
  if (i < n) out[meta[metaIdx] + i] = (float)mask[i];
}

// ---------------------------------------------------------------------------
extern "C" void kernel_launch(void* const* d_in, const int* in_sizes, int n_in,
                              void* d_out, int out_size, void* d_ws, size_t ws_size,
                              hipStream_t stream) {
  (void)in_sizes; (void)n_in; (void)out_size; (void)ws_size;
  const float* images = (const float*)d_in[0];
  float* out = (float*)d_out;

  // workspace layout (~19 MB)
  float* img1  = (float*)d_ws;            // 8*80^3  = 4,096,000 f32
  float* img2  = img1 + 4096000;          // 8*40^3  =   512,000 f32
  int*   mask0 = (int*)(img2 + 512000);   // 64000
  int*   mask1 = mask0 + NP0;             // 8000
  int*   mask2 = mask1 + NP1;             // 1000
  int*   off0  = mask2 + NP2;             // 64000
  int*   off1  = off0 + NP0;              // 8000
  int*   off2  = off1 + NP1;              // 1000
  int*   meta  = off2 + NP2;              // 12 ints

  // thrSS = thr^2 * (n-1), n = 512 (ddof=1)
  const float thr0 = 1.000f * 1.000f * 511.0f;
  const float thr1 = 0.354f * 0.354f * 511.0f;
  const float thr2 = 0.125f * 0.125f * 511.0f;

  down2_kernel<<<(4096000 + 255) / 256, 256, 0, stream>>>(images, img1);
  down4_kernel<<<(512000  + 255) / 256, 256, 0, stream>>>(images, img2);

  patch_std_kernel<160><<<NP0 / 16,        32, 0, stream>>>(images, mask0, thr0, NP0);
  patch_std_kernel< 80><<<NP1 / 16,        32, 0, stream>>>(img1,   mask1, thr1, NP1);
  patch_std_kernel< 40><<<(NP2 + 15) / 16, 32, 0, stream>>>(img2,   mask2, thr2, NP2);

  scan_kernel<<<1, 1024, 0, stream>>>(mask0, off0, NP0);
  scan_kernel<<<1, 1024, 0, stream>>>(mask1, off1, NP1);
  scan_kernel<<<1, 1024, 0, stream>>>(mask2, off2, NP2);

  meta_kernel<<<1, 256, 0, stream>>>(mask0, off0, mask1, off1, mask2, off2, meta, out);

  gather_resized<160><<<NP0, 64, 0, stream>>>(images, mask0, off0, meta, 0, out);
  gather_resized< 80><<<NP1, 64, 0, stream>>>(img1,   mask1, off1, meta, 1, out);
  gather_resized< 40><<<NP2, 64, 0, stream>>>(img2,   mask2, off2, meta, 2, out);

  gather_full<16><<<NP1, 256, 0, stream>>>(images, mask1, off1, meta, 3, out);
  gather_full<32><<<NP2, 256, 0, stream>>>(images, mask2, off2, meta, 4, out);

  posmask_kernel<<<(NP0 + 255) / 256, 256, 0, stream>>>(mask0, NP0, meta, 5, out);
  posmask_kernel<<<(NP1 + 255) / 256, 256, 0, stream>>>(mask1, NP1, meta, 6, out);
  posmask_kernel<<<(NP2 + 255) / 256, 256, 0, stream>>>(mask2, NP2, meta, 7, out);
}
